// SparseMoeBlock_506806141322
// MI455X (gfx1250) — compile-verified, matrix operand order checked by reading, not run on hardware
//
#include <hip/hip_runtime.h>
#include <hip/hip_bf16.h>
#include <math.h>

// Problem constants (from reference)
#define NTOK 8192      // B*S
#define DMODEL 2048
#define NEXP 8
#define DFF 8192
#define TOPK 2

typedef __attribute__((ext_vector_type(16))) __bf16 v16bf;
typedef __attribute__((ext_vector_type(8)))  __bf16 v8bf;
typedef __attribute__((ext_vector_type(8)))  float  v8f;
typedef __attribute__((ext_vector_type(4)))  int    v4i;

#define GAS __attribute__((address_space(1)))
#define LAS __attribute__((address_space(3)))

// ---------------- workspace layout (bytes) ----------------
#define OFF_LOGITS 0
#define OFF_SEL    0x40000
#define OFF_XBF    0x41000
#define OFF_W1T    (OFF_XBF + (size_t)NTOK * DMODEL * 2)          // [DFF][DMODEL] bf16
#define OFF_W2T    (OFF_W1T + (size_t)DFF * DMODEL * 2)           // [DMODEL][DFF] bf16
#define OFF_H      (OFF_W2T + (size_t)DMODEL * DFF * 2)           // [NTOK][DFF]  bf16

// ---------------- CDNA5 async global->LDS staging ----------------
#if __has_builtin(__builtin_amdgcn_global_load_async_to_lds_b128)
#define HAVE_ASYNC_B128 1
#else
#define HAVE_ASYNC_B128 0
#endif

__device__ __forceinline__ void stage16(__bf16* dst_lds, const __bf16* src_glb) {
#if HAVE_ASYNC_B128
    __builtin_amdgcn_global_load_async_to_lds_b128(
        (GAS v4i*)src_glb,     // drops const + addrspacecast to AS1
        (LAS v4i*)dst_lds,     // addrspacecast to AS3
        0, 0);
#else
    *(uint4*)dst_lds = *(const uint4*)src_glb;
#endif
}

// wait until ASYNCcnt <= 8 (previous tile's 8 per-thread loads done,
// next tile's 8 may still be in flight)
__device__ __forceinline__ void async_wait_le8() {
#if HAVE_ASYNC_B128
#if __has_builtin(__builtin_amdgcn_s_wait_asynccnt)
    __builtin_amdgcn_s_wait_asynccnt(8);
#else
    asm volatile("s_wait_asynccnt 0x8" ::: "memory");
#endif
#endif
}

__device__ __forceinline__ void async_wait_le0() {
#if HAVE_ASYNC_B128
#if __has_builtin(__builtin_amdgcn_s_wait_asynccnt)
    __builtin_amdgcn_s_wait_asynccnt(0);
#else
    asm volatile("s_wait_asynccnt 0x0" ::: "memory");
#endif
#endif
}

// =====================================================================
// Kernel 1: router logits + fp32 -> bf16 conversion of x
// =====================================================================
__global__ void __launch_bounds__(256) moe_router_kernel(
        const float* __restrict__ x,
        const float* __restrict__ gate_w,
        float* __restrict__ logits,
        __bf16* __restrict__ xbf) {
    const int n   = blockIdx.x;
    const int tid = threadIdx.x;
    float acc[NEXP];
#pragma unroll
    for (int e = 0; e < NEXP; ++e) acc[e] = 0.0f;

    const size_t rowoff = (size_t)n * DMODEL;
    for (int d = tid; d < DMODEL; d += 256) {
        float xv = x[rowoff + d];
        xbf[rowoff + d] = (__bf16)xv;
#pragma unroll
        for (int e = 0; e < NEXP; ++e)
            acc[e] += xv * gate_w[e * DMODEL + d];
    }

    __shared__ float red[256][NEXP];
#pragma unroll
    for (int e = 0; e < NEXP; ++e) red[tid][e] = acc[e];
    __syncthreads();
    for (int off = 128; off > 0; off >>= 1) {
        if (tid < off) {
#pragma unroll
            for (int e = 0; e < NEXP; ++e) red[tid][e] += red[tid + off][e];
        }
        __syncthreads();
    }
    if (tid < NEXP) logits[(size_t)n * NEXP + tid] = red[0][tid];
}

// =====================================================================
// Kernel 2: deterministic logit sum over tokens + top-2 selection
// =====================================================================
__global__ void __launch_bounds__(256) moe_select_kernel(
        const float* __restrict__ logits, int* __restrict__ sel) {
    const int tid = threadIdx.x;
    __shared__ float part[256][NEXP];
    float acc[NEXP];
#pragma unroll
    for (int e = 0; e < NEXP; ++e) acc[e] = 0.0f;
    for (int n = tid; n < NTOK; n += 256) {
#pragma unroll
        for (int e = 0; e < NEXP; ++e) acc[e] += logits[(size_t)n * NEXP + e];
    }
#pragma unroll
    for (int e = 0; e < NEXP; ++e) part[tid][e] = acc[e];
    __syncthreads();
    for (int off = 128; off > 0; off >>= 1) {
        if (tid < off) {
#pragma unroll
            for (int e = 0; e < NEXP; ++e) part[tid][e] += part[tid + off][e];
        }
        __syncthreads();
    }
    if (tid == 0) {
        float best = -INFINITY; int bi = 0;
        for (int e = 0; e < NEXP; ++e)
            if (part[0][e] > best) { best = part[0][e]; bi = e; }
        float best2 = -INFINITY; int bi2 = 0;
        for (int e = 0; e < NEXP; ++e)
            if (e != bi && part[0][e] > best2) { best2 = part[0][e]; bi2 = e; }
        sel[0] = bi;
        sel[1] = bi2;
    }
}

// =====================================================================
// Kernel 3: transpose + convert selected expert weight matrix
//   W[e] f32 [Krows][Ncols]  ->  outT bf16 [Ncols][Krows]
// =====================================================================
#define TCK 72   // LDS row stride in halves
__global__ void __launch_bounds__(256) moe_convT_kernel(
        const float* __restrict__ W, const int* __restrict__ sel, int kIdx,
        int Krows, int Ncols, __bf16* __restrict__ outT) {
    __shared__ __align__(16) __bf16 t[64 * TCK];   // [n][k]
    const int tid = threadIdx.x;
    const int n0 = blockIdx.x * 64;
    const int k0 = blockIdx.y * 64;
    const int e  = sel[kIdx];
    const float* We = W + (size_t)e * Krows * Ncols;

#pragma unroll
    for (int i = 0; i < 16; ++i) {
        int j = tid + i * 256;             // 0..4095
        int k = j >> 6;
        int n = j & 63;
        t[n * TCK + k] = (__bf16)We[(size_t)(k0 + k) * Ncols + n0 + n];
    }
    __syncthreads();
#pragma unroll
    for (int i = 0; i < 2; ++i) {
        int idx = tid + i * 256;           // 0..511
        int n = idx >> 3;
        int c = idx & 7;
        *(uint4*)(&outT[(size_t)(n0 + n) * Krows + k0 + c * 8]) =
            *(const uint4*)(&t[n * TCK + c * 8]);
    }
}

// =====================================================================
// helpers
// =====================================================================
__device__ __forceinline__ v16bf load_frag16(const __bf16* p) {
    v8bf lo = *(const v8bf*)(p);
    v8bf hi = *(const v8bf*)(p + 8);
    return __builtin_shufflevector(lo, hi, 0, 1, 2, 3, 4, 5, 6, 7,
                                   8, 9, 10, 11, 12, 13, 14, 15);
}

// branchless gelu (tanh approx): tanh(u) = 2/(1+exp(-2u)) - 1
__device__ __forceinline__ float gelu_tanh(float v) {
    float u  = 0.7978845608028654f * (v + 0.044715f * v * v * v);
    float th = 2.0f * __builtin_amdgcn_rcpf(1.0f + __expf(-2.0f * u)) - 1.0f;
    return 0.5f * v * (1.0f + th);
}

// =====================================================================
// Tiled bf16 WMMA GEMM, 128x128 block tile, K-step 64, 256 threads,
// DOUBLE-BUFFERED async global->LDS staging.
//   A  : [M][Kdim] bf16 row-major (x or h)
//   BT : [Ntot][Kdim] bf16 row-major (pre-transposed expert weights)
//   MODE 0: outH = gelu(A @ B + bias[e])                  (bf16)
//   MODE 1: outF (=|+=) rw[token] * (A @ B + bias[e])     (f32)
// =====================================================================
#define LDS_K 72                           // row stride in halves
#define TILE_BYTES (128 * LDS_K * 2)       // 18432 B per A/B tile
#define SMEM_BYTES (4 * TILE_BYTES)        // 73728 B (2 x (A+B)); epilogue reuses

__device__ __forceinline__ void stage_tile(
        __bf16* la, __bf16* lb,
        const __bf16* __restrict__ A, const __bf16* __restrict__ BT,
        int rowBase, int colBase, int k0, int Kdim, int tid) {
#pragma unroll
    for (int i = 0; i < 4; ++i) {
        int idx = tid + i * 256;           // 0..1023
        int r = idx >> 3;                  // row 0..127
        int c = idx & 7;                   // 8-half chunk
        stage16(&la[r * LDS_K + c * 8],
                A + (size_t)(rowBase + r) * Kdim + k0 + c * 8);
        stage16(&lb[r * LDS_K + c * 8],
                BT + (size_t)(colBase + r) * Kdim + k0 + c * 8);
    }
}

template <int MODE>
__global__ void __launch_bounds__(256, 2) moe_gemm_kernel(
        const __bf16* __restrict__ A,
        const __bf16* __restrict__ BT,
        const float* __restrict__ bias,
        const int* __restrict__ sel,
        int kIdx, int Kdim, int Ntot,
        __bf16* __restrict__ outH,
        float* __restrict__ outF,
        const float* __restrict__ logits,
        int accumFlag) {
    __shared__ __align__(16) unsigned char smem[SMEM_BYTES];
    __shared__ float lds_rw[128];
    __shared__ float lds_bias[128];

    __bf16* bufA0 = (__bf16*)(smem);
    __bf16* bufB0 = (__bf16*)(smem + TILE_BYTES);
    __bf16* bufA1 = (__bf16*)(smem + 2 * TILE_BYTES);
    __bf16* bufB1 = (__bf16*)(smem + 3 * TILE_BYTES);

    const int tid    = threadIdx.x;
    const int lane   = tid & 31;
    const int wave   = tid >> 5;       // 0..7
    const int wave_m = wave & 3;       // 4 row-groups of 32
    const int wave_n = wave >> 2;      // 2 col-groups of 64
    const int lane15 = lane & 15;
    const int khalf  = (lane >> 4) << 4;   // 0 or 16

    const int colBase = blockIdx.x * 128;
    const int rowBase = blockIdx.y * 128;

    const int e = sel[kIdx];
    const float* be = bias + (size_t)e * Ntot;

    if (tid < 128) {
        lds_bias[tid] = be[colBase + tid];
        if (MODE == 1) {
            int n = rowBase + tid;
            float l0 = logits[(size_t)n * NEXP + sel[0]];
            float l1 = logits[(size_t)n * NEXP + sel[1]];
            float mx = fmaxf(l0, l1);
            float e0 = __expf(l0 - mx), e1 = __expf(l1 - mx);
            lds_rw[tid] = (kIdx == 0 ? e0 : e1) *
                          __builtin_amdgcn_rcpf(e0 + e1);
        }
    }

    v8f acc[2][4] = {};

    // prologue: issue tile 0 into buffer 0
    stage_tile(bufA0, bufB0, A, BT, rowBase, colBase, 0, Kdim, tid);

    for (int k0 = 0; k0 < Kdim; k0 += 64) {
        const int cur = (k0 >> 6) & 1;
        __bf16* la = cur ? bufA1 : bufA0;
        __bf16* lb = cur ? bufB1 : bufB0;
        __bf16* na = cur ? bufA0 : bufA1;
        __bf16* nb = cur ? bufB0 : bufB1;

        // all waves finished reading the buffer we are about to overwrite
        __syncthreads();

        const bool has_next = (k0 + 64) < Kdim;
        if (has_next) {
            stage_tile(na, nb, A, BT, rowBase, colBase, k0 + 64, Kdim, tid);
            async_wait_le8();   // current tile landed; next still in flight
        } else {
            async_wait_le0();   // drain everything
        }
        __syncthreads();        // current tile visible to all waves

        // ---- two K=32 sub-steps; 8 WMMAs each per wave ----
#pragma unroll
        for (int kk = 0; kk < 64; kk += 32) {
            v16bf af[2];
#pragma unroll
            for (int ti = 0; ti < 2; ++ti)
                af[ti] = load_frag16(
                    &la[(wave_m * 32 + ti * 16 + lane15) * LDS_K + kk + khalf]);
            v16bf bfm[4];
#pragma unroll
            for (int tj = 0; tj < 4; ++tj)
                bfm[tj] = load_frag16(
                    &lb[(wave_n * 64 + tj * 16 + lane15) * LDS_K + kk + khalf]);

#pragma unroll
            for (int ti = 0; ti < 2; ++ti)
#pragma unroll
                for (int tj = 0; tj < 4; ++tj)
                    acc[ti][tj] = __builtin_amdgcn_wmma_f32_16x16x32_bf16(
                        false, af[ti], false, bfm[tj], (short)0, acc[ti][tj],
                        false, false);
        }
    }

    // ===== epilogue: stage C through LDS (reusing smem), vector stores =====
    const int mofs = (lane >> 4) << 3;     // +0 or +8 row inside 16-tile
    __bf16* sth = (__bf16*)smem;           // [64][136] halves  (MODE 0)
    float*  stf = (float*)smem;            // [64][132] floats  (MODE 1)

#pragma unroll
    for (int p = 0; p < 2; ++p) {          // two 64-row half-tiles
        __syncthreads();
        if ((wave_m >> 1) == p) {
#pragma unroll
            for (int ti = 0; ti < 2; ++ti) {
#pragma unroll
                for (int tj = 0; tj < 4; ++tj) {
                    int lcol = wave_n * 64 + tj * 16 + lane15;
                    float bv = lds_bias[lcol];
#pragma unroll
                    for (int r = 0; r < 8; ++r) {
                        int hrow = (wave_m & 1) * 32 + ti * 16 + r + mofs; // 0..63
                        float v = acc[ti][tj][r] + bv;
                        if (MODE == 0) {
                            sth[hrow * 136 + lcol] = (__bf16)gelu_tanh(v);
                        } else {
                            stf[hrow * 132 + lcol] = lds_rw[p * 64 + hrow] * v;
                        }
                    }
                }
            }
        }
        __syncthreads();

        if (MODE == 0) {
#pragma unroll
            for (int i = 0; i < 4; ++i) {
                int idx = tid + i * 256;   // 0..1023
                int r = idx >> 4;          // 0..63
                int c = idx & 15;          // 8-half chunk
                *(uint4*)(&outH[(size_t)(rowBase + p * 64 + r) * Ntot +
                                colBase + c * 8]) =
                    *(const uint4*)(&sth[r * 136 + c * 8]);
            }
        } else {
#pragma unroll
            for (int i = 0; i < 8; ++i) {
                int idx = tid + i * 256;   // 0..2047
                int r = idx >> 5;          // 0..63
                int c = idx & 31;          // 4-float chunk
                size_t o = (size_t)(rowBase + p * 64 + r) * Ntot + colBase + c * 4;
                const float* sp = &stf[r * 132 + c * 4];
                float v0 = sp[0], v1 = sp[1], v2 = sp[2], v3 = sp[3];
                if (accumFlag) {
                    float4 ov = *(const float4*)&outF[o];
                    v0 += ov.x; v1 += ov.y; v2 += ov.z; v3 += ov.w;
                }
                float4 res; res.x = v0; res.y = v1; res.z = v2; res.w = v3;
                *(float4*)&outF[o] = res;
            }
        }
    }
}

// =====================================================================
// Host launcher
// =====================================================================
extern "C" void kernel_launch(void* const* d_in, const int* in_sizes, int n_in,
                              void* d_out, int out_size, void* d_ws, size_t ws_size,
                              hipStream_t stream) {
    const float* x      = (const float*)d_in[0];   // [B,S,D]
    const float* gate_w = (const float*)d_in[1];   // [E,D]
    const float* w1     = (const float*)d_in[2];   // [E,D,DFF]
    const float* b1     = (const float*)d_in[3];   // [E,DFF]
    const float* w2     = (const float*)d_in[4];   // [E,DFF,D]
    const float* b2     = (const float*)d_in[5];   // [E,D]
    float* out          = (float*)d_out;           // [B,S,D]

    char* ws = (char*)d_ws;
    float*  logits = (float*)(ws + OFF_LOGITS);
    int*    sel    = (int*)(ws + OFF_SEL);
    __bf16* xbf    = (__bf16*)(ws + OFF_XBF);
    __bf16* w1t    = (__bf16*)(ws + OFF_W1T);
    __bf16* w2t    = (__bf16*)(ws + OFF_W2T);
    __bf16* hbf    = (__bf16*)(ws + OFF_H);

    moe_router_kernel<<<NTOK, 256, 0, stream>>>(x, gate_w, logits, xbf);
    moe_select_kernel<<<1, 256, 0, stream>>>(logits, sel);

    dim3 gridC1(DFF / 64, DMODEL / 64);
    dim3 gridC2(DMODEL / 64, DFF / 64);
    dim3 grid1(DFF / 128, NTOK / 128);
    dim3 grid2(DMODEL / 128, NTOK / 128);

    for (int k = 0; k < TOPK; ++k) {
        moe_convT_kernel<<<gridC1, 256, 0, stream>>>(w1, sel, k, DMODEL, DFF, w1t);
        moe_gemm_kernel<0><<<grid1, 256, 0, stream>>>(
            xbf, w1t, b1, sel, k, DMODEL, DFF, hbf, nullptr, nullptr, 0);
        moe_convT_kernel<<<gridC2, 256, 0, stream>>>(w2, sel, k, DFF, DMODEL, w2t);
        moe_gemm_kernel<1><<<grid2, 256, 0, stream>>>(
            hbf, w2t, b2, sel, k, DFF, DMODEL, nullptr, out, logits, k);
    }
}
